// ProtocolTreeAttention_32804960207331
// MI455X (gfx1250) — matrix operand order repaired and checked by reading, not compiled, not run on hardware
//
#include <hip/hip_runtime.h>
#include <hip/hip_bf16.h>
#include <math.h>

// ---------------------------------------------------------------------------
// Types for CDNA5 WMMA (wave32): v_wmma_f32_16x16x32_bf16
// ---------------------------------------------------------------------------
typedef __bf16 bf16;
typedef __attribute__((ext_vector_type(8)))  bf16  v8bf;
typedef __attribute__((ext_vector_type(16))) bf16  v16bf;
typedef __attribute__((ext_vector_type(8)))  float v8f;

// Global (addrspace 1) pointer views: force global_load_* (not flat_load_*),
// keeping weight traffic on LOADcnt only (flat would also tie up DScnt).
typedef const v16bf __attribute__((address_space(1)))* gv16bf_p;
typedef const float __attribute__((address_space(1)))* gf32_p;

#define NWAVES 4
#define BLOCK  (NWAVES * 32)

__device__ __forceinline__ float gelu_exact(float x) {
    return 0.5f * x * (1.0f + erff(x * 0.70710678118654752f));
}

// ---------------------------------------------------------------------------
// WMMA GEMM: D[M x NC] = A[M x Kp] * W^T + bias   (cooperative: NWAVES waves
// split the N-tile space with compile-time uniform trip counts; caller must
// __syncthreads() before consuming D).
//   A  : LDS bf16, row-major, row stride Kp (multiple of 32)
//   W  : global bf16, row-major (NC x Kp)  -> B operand = W^T, loaded natively
//   M, NC multiples of 16; NC multiple of 64. ACT: 0 bf16, 1 gelu->bf16, 2 f32
// ---------------------------------------------------------------------------
template<int ACT, int Kp, int M, int NC>
__device__ void gemm_wmma(const bf16* __restrict__ A,
                          const bf16* __restrict__ W, const float* __restrict__ bias,
                          bf16* Dbf, float* Df, int ldd) {
    static_assert(NC % (NWAVES * 16) == 0, "NC must be a multiple of 64");
    constexpr int NT = NC / (NWAVES * 16);      // nt tiles per wave (uniform)
    const int lane = threadIdx.x & 31;
    const int wave = threadIdx.x >> 5;
    const int am   = lane & 15;          // A: row M
    const int akb  = (lane >> 4) << 4;   // A: K base 0 / 16 (matches B layout)
    const int bn   = lane & 15;          // B: col N
    const int bkl  = (lane >> 4) << 4;   // B: K base 0 / 16 (contiguous k order, same as A)
    const int mb   = (lane >> 4) << 3;   // C/D: row base 0 / 8

    for (int mt = 0; mt < M; mt += 16) {
        const bf16* Arow = A + (size_t)(mt + am) * Kp;
        for (int i = 0; i < NT; ++i) {
            const int nt = (wave + i * NWAVES) * 16;
            v8f c = {0.f,0.f,0.f,0.f,0.f,0.f,0.f,0.f};
            const bf16* Wrow = W + (size_t)(nt + bn) * Kp + bkl;
#pragma unroll
            for (int k = 0; k < Kp; k += 32) {
                v8bf a0 = *(const v8bf*)(Arow + k + akb);        // K = kb+0..7
                v8bf a1 = *(const v8bf*)(Arow + k + akb + 8);    // K = kb+8..15
                v16bf a = __builtin_shufflevector(a0, a1,
                            0,1,2,3,4,5,6,7,8,9,10,11,12,13,14,15);
                v16bf b = *(gv16bf_p)(Wrow + k);                 // global_load_b128 x2
                c = __builtin_amdgcn_wmma_f32_16x16x32_bf16(
                        false, a, false, b, (short)0, c, false, false);
            }
            const float bv = *(gf32_p)(bias + nt + bn);          // global_load_b32
#pragma unroll
            for (int r = 0; r < 8; ++r) {
                float v = c[r] + bv;
                if constexpr (ACT == 1) v = gelu_exact(v);
                const int row = mt + mb + r;
                if constexpr (ACT == 2) Df[(size_t)row * ldd + nt + bn] = v;
                else                    Dbf[(size_t)row * ldd + nt + bn] = (bf16)v;
            }
        }
    }
}

// ---------------------------------------------------------------------------
// CLS-query attention. kv: LDS (16*N rows, token-major row = (t-1)*16 + p),
// row = [k(0..E-1) | v(E..2E-1)]. qkvcls: f32 [q|k|v] of the constant CLS row.
// Writes o_bf[p*E + e] (bf16), 16 packets. One (packet,head) per thread (64).
// ---------------------------------------------------------------------------
template<int E, int N>
__device__ void attention_cls(const float* __restrict__ qkvcls,
                              const bf16* __restrict__ kv,
                              bf16* __restrict__ o_bf) {
    constexpr int H = 4, Dh = E / H, T = N + 1;
    const float scale = rsqrtf((float)Dh);
    const int tid = threadIdx.x;
    if (tid < 64) {
        const int p = tid >> 2, h = tid & 3;
        const float* q  = qkvcls + h * Dh;
        const float* kc = qkvcls + E     + h * Dh;
        const float* vc = qkvcls + 2 * E + h * Dh;
        float s[T];
        float m = -3.0e38f;
#pragma unroll
        for (int t = 0; t < T; ++t) {
            float acc = 0.f;
            if (t == 0) {
                for (int d = 0; d < Dh; ++d) acc += q[d] * kc[d];
            } else {
                const bf16* kr = kv + (size_t)((t - 1) * 16 + p) * (2 * E) + h * Dh;
                for (int d = 0; d < Dh; ++d) acc += q[d] * (float)kr[d];
            }
            s[t] = acc * scale;
            m = fmaxf(m, s[t]);
        }
        float den = 0.f;
#pragma unroll
        for (int t = 0; t < T; ++t) { s[t] = expf(s[t] - m); den += s[t]; }
        const float inv = 1.f / den;
        for (int d = 0; d < Dh; ++d) {
            float acc = s[0] * vc[d];
#pragma unroll
            for (int t = 1; t < T; ++t) {
                const bf16* vr = kv + (size_t)((t - 1) * 16 + p) * (2 * E) + E + h * Dh;
                acc += s[t] * (float)vr[d];
            }
            o_bf[p * E + h * Dh + d] = (bf16)(acc * inv);
        }
    }
}

// ---------------------------------------------------------------------------
// LayerNorm over 16 rows of width E: y = g*((a+b)-mu)/sqrt(var+eps)+beta.
// a: stride sa (0 => broadcast, e.g. the constant cls row). In-place safe.
// ---------------------------------------------------------------------------
template<int E>
__device__ void ln16(const float* __restrict__ A, int sa,
                     const float* __restrict__ B, int sb,
                     const float* __restrict__ g, const float* __restrict__ beta,
                     float* Of, int sof, bf16* Ob, int sob) {
    const int tid = threadIdx.x;
    if (tid < 16) {
        const float* ra = A + (size_t)tid * sa;
        const float* rb = B + (size_t)tid * sb;
        float mu = 0.f;
        for (int e = 0; e < E; ++e) mu += ra[e] + rb[e];
        mu *= (1.f / E);
        float var = 0.f;
        for (int e = 0; e < E; ++e) { float d = ra[e] + rb[e] - mu; var += d * d; }
        var *= (1.f / E);
        const float inv = rsqrtf(var + 1e-5f);
        for (int e = 0; e < E; ++e) {
            float y = g[e] * (ra[e] + rb[e] - mu) * inv + beta[e];
            if (Of) Of[(size_t)tid * sof + e] = y;
            if (Ob) Ob[(size_t)tid * sob + e] = (bf16)y;
        }
    }
}

// ---------------------------------------------------------------------------
// Aggregator (CLS output only): tok = 16*N rows x E (token-major), out 16 x E.
// ---------------------------------------------------------------------------
struct AggP {
    const bf16  *Wqkv, *Wo, *W1, *W2;                 // bf16 (ws)
    const float *bqkv, *bo, *b1, *b2, *ln_g, *ln_b, *cls, *qkvcls;  // f32
};

template<int E, int N>
__device__ void run_agg(const AggP& ap, const bf16* tok,
                        char* tokzone, char* kvzone, bf16* outv) {
    constexpr int R = 16 * N;
    bf16* kv = (bf16*)kvzone;                                   // R x 2E
    // K,V for real tokens (rows E..3E-1 of Wqkv)
    gemm_wmma<0, E, R, 2 * E>(tok, ap.Wqkv + (size_t)E * E, ap.bqkv + E, kv, nullptr, 2 * E);
    __syncthreads();
    bf16*  o_bf = (bf16*)tokzone;                               // 16 x E   (tok dead)
    attention_cls<E, N>(ap.qkvcls, kv, o_bf);
    __syncthreads();
    float* h_f  = (float*)(tokzone + 4096);                     // 16 x E f32
    bf16*  h_bf = (bf16*)(tokzone + 12288);                     // 16 x E
    gemm_wmma<2, E, 16, E>(o_bf, ap.Wo, ap.bo, nullptr, h_f, E);
    __syncthreads();
    // h = LN(cls + o@Wo^T + bo)
    ln16<E>(ap.cls, 0, h_f, E, ap.ln_g, ap.ln_b, h_f, E, h_bf, E);
    __syncthreads();
    bf16*  G   = (bf16*)kvzone;                                 // 16 x 4E  (kv dead)
    float* f_f = (float*)(kvzone + 16384);                      // 16 x E f32
    gemm_wmma<1, E, 16, 4 * E>(h_bf, ap.W1, ap.b1, G, nullptr, 4 * E);
    __syncthreads();
    gemm_wmma<2, 4 * E, 16, E>(G, ap.W2, ap.b2, nullptr, f_f, E);
    __syncthreads();
    ln16<E>(h_f, E, f_f, E, ap.ln_g, ap.ln_b, nullptr, 0, outv, E);
    __syncthreads();
}

// ---------------------------------------------------------------------------
// Kernel arguments (resolved device pointers, passed by value)
// ---------------------------------------------------------------------------
struct MainArgs {
    const float* x;
    float*       out;
    const bf16*  subW[7]; const float* subB[7]; int subOff[7];  // ip.flags(3)+tcp.flags(4)
    AggP         subAgg[2];
    const bf16*  fW[15];  const float* fB[15];  int fOff[15];   // eth(3)+ip(6)+tcp(6); -1 => upgraded
    AggP         protoAgg[3];
    AggP         finalAgg;
    const float* clfW; const float* clfB;
};

#define TOKZONE_BYTES 24576
#define KVZONE_BYTES  49152
#define SMEM_BYTES    (TOKZONE_BYTES + KVZONE_BYTES + 4096 + 12288 + 4096)

__global__ __launch_bounds__(BLOCK)
void k_fused(MainArgs P) {
    extern __shared__ char smem[];
    char* tokzone = smem;
    char* kvzone  = smem + TOKZONE_BYTES;
    bf16* upg     = (bf16*)(smem + TOKZONE_BYTES + KVZONE_BYTES);   // 2 x (16x64)
    bf16* protov  = upg + 2 * 16 * 64;                              // 3 x (16x128)
    bf16* packet  = protov + 3 * 16 * 128;                          // 16 x 128
    const int tid = threadIdx.x;
    const size_t gbase = (size_t)blockIdx.x * 16;

    // ---------- Stage 1: sub-aggregators (E=64) over 16-dim flag subfields ----
    {
        const int subN[2] = {3, 4};
        int tb = 0;
        for (int s = 0; s < 2; ++s) {
            const int N = subN[s], R = 16 * N;
            bf16* A_in = (bf16*)kvzone;      // R x 32 (K padded 16->32)
            bf16* tok  = (bf16*)tokzone;     // R x 64
            for (int i = tid; i < R * 32; i += BLOCK) {
                int row = i >> 5, k = i & 31, t = row >> 4, p = row & 15;
                float v = (k < 16) ? P.x[(gbase + p) * 976 + P.subOff[tb + t] + k] : 0.f;
                A_in[i] = (bf16)v;
            }
            __syncthreads();
            for (int t = 0; t < N; ++t)
                gemm_wmma<0, 32, 16, 64>(A_in + t * 16 * 32, P.subW[tb + t], P.subB[tb + t],
                                         tok + t * 16 * 64, nullptr, 64);
            __syncthreads();
            if (s == 0) run_agg<64, 3>(P.subAgg[0], tok, tokzone, kvzone, upg);
            else        run_agg<64, 4>(P.subAgg[1], tok, tokzone, kvzone, upg + 16 * 64);
            tb += N;
        }
    }

    // ---------- Stage 2: field aligners + proto aggregators (E=128) ----------
    {
        const int protoN[3] = {3, 6, 6};
        int fb = 0;
        for (int pr = 0; pr < 3; ++pr) {
            const int N = protoN[pr], R = 16 * N;
            bf16* A_in = (bf16*)kvzone;      // R x 64
            bf16* tok  = (bf16*)tokzone;     // R x 128
            for (int i = tid; i < R * 64; i += BLOCK) {
                int row = i >> 6, k = i & 63, t = row >> 4, p = row & 15;
                int off = P.fOff[fb + t];
                float v;
                if (off >= 0) v = P.x[(gbase + p) * 976 + off + k];
                else          v = (float)upg[((pr == 1) ? 0 : 1) * 16 * 64 + p * 64 + k];
                A_in[i] = (bf16)v;
            }
            __syncthreads();
            for (int t = 0; t < N; ++t)
                gemm_wmma<0, 64, 16, 128>(A_in + t * 16 * 64, P.fW[fb + t], P.fB[fb + t],
                                          tok + t * 16 * 128, nullptr, 128);
            __syncthreads();
            if      (pr == 0) run_agg<128, 3>(P.protoAgg[0], tok, tokzone, kvzone, protov);
            else if (pr == 1) run_agg<128, 6>(P.protoAgg[1], tok, tokzone, kvzone, protov + 16 * 128);
            else              run_agg<128, 6>(P.protoAgg[2], tok, tokzone, kvzone, protov + 2 * 16 * 128);
            fb += N;
        }
    }

    // ---------- Stage 3: final aggregator over the 3 proto vectors ----------
    run_agg<128, 3>(P.finalAgg, protov, tokzone, kvzone, packet);

    // ---------- Stage 4: classifier (2 outputs) ----------
    if (tid < 16) {
        const int p = tid;
        float a0 = P.clfB[0], a1 = P.clfB[1];
        for (int e = 0; e < 128; ++e) {
            float v = (float)packet[p * 128 + e];
            a0 += v * P.clfW[e];
            a1 += v * P.clfW[128 + e];
        }
        P.out[(gbase + p) * 2 + 0] = a0;
        P.out[(gbase + p) * 2 + 1] = a1;
    }
}

// ---------------------------------------------------------------------------
// Prep kernel 1: fp32 weights -> bf16 workspace (K zero-padded to mult of 32)
// ---------------------------------------------------------------------------
struct WConv { const float* src; unsigned int dstOff; int N, K, Kp; };
struct WConvTab { WConv w[46]; };

__global__ void k_convert_weights(WConvTab t, bf16* wsbf) {
    const WConv w = t.w[blockIdx.x];
    const int total = w.N * w.Kp;
    for (int i = threadIdx.x; i < total; i += blockDim.x) {
        int r = i / w.Kp, k = i - r * w.Kp;
        float v = (k < w.K) ? w.src[(size_t)r * w.K + k] : 0.f;
        wsbf[w.dstOff + i] = (bf16)v;
    }
}

// ---------------------------------------------------------------------------
// Prep kernel 2: qkvcls = Wqkv @ cls + bqkv  (f32, per aggregator, 3E values)
// ---------------------------------------------------------------------------
struct AggCls { const float* Wqkv; const float* bqkv; const float* cls; int E; int off; };
struct AggClsTab { AggCls a[6]; };

__global__ void k_qkvcls(AggClsTab t, float* qc) {
    const AggCls a = t.a[blockIdx.x];
    for (int j = threadIdx.x; j < 3 * a.E; j += blockDim.x) {
        float s = a.bqkv[j];
        for (int k = 0; k < a.E; ++k) s += a.Wqkv[(size_t)j * a.E + k] * a.cls[k];
        qc[a.off + j] = s;
    }
}

// ---------------------------------------------------------------------------
// Host side. Input flattening assumed = jax pytree order over setup_inputs():
// top-level dict order {x, sub_params, field_aligners, proto_aggs, final_agg,
// clf_W, clf_b}; nested dicts alphabetical, tuples (W, b) in order. 113 leaves.
// ---------------------------------------------------------------------------
struct WInfo { int idx, N, K, Kp; };
static const WInfo kWTab[46] = {
    // sub ip.flags agg: W1, W2, Wo, Wqkv
    {1,256,64,64},{2,64,256,256},{3,64,64,64},{4,192,64,64},
    // sub ip.flags aligners (df, mf, rb)
    {12,64,16,32},{14,64,16,32},{16,64,16,32},
    // sub tcp.flags agg
    {18,256,64,64},{19,64,256,256},{20,64,64,64},{21,192,64,64},
    // sub tcp.flags aligners (ack, fin, rst, syn)
    {29,64,16,32},{31,64,16,32},{33,64,16,32},{35,64,16,32},
    // field aligners (alphabetical)
    {37,128,64,64},{39,128,64,64},{41,128,64,64},{43,128,64,64},{45,128,64,64},
    {47,128,64,64},{49,128,64,64},{51,128,64,64},{53,128,64,64},{55,128,64,64},
    {57,128,64,64},{59,128,64,64},{61,128,64,64},{63,128,64,64},{65,128,64,64},
    // proto aggs eth/ip/tcp + final: W1, W2, Wo, Wqkv
    {67,512,128,128},{68,128,512,512},{69,128,128,128},{70,384,128,128},
    {78,512,128,128},{79,128,512,512},{80,128,128,128},{81,384,128,128},
    {89,512,128,128},{90,128,512,512},{91,128,128,128},{92,384,128,128},
    {100,512,128,128},{101,128,512,512},{102,128,128,128},{103,384,128,128},
};

extern "C" void kernel_launch(void* const* d_in, const int* in_sizes, int n_in,
                              void* d_out, int out_size, void* d_ws, size_t ws_size,
                              hipStream_t stream) {
    (void)n_in; (void)out_size; (void)ws_size;
    const float* const* fin = (const float* const*)d_in;
    bf16* wsbf = (bf16*)d_ws;

    // --- bf16 weight workspace layout ---
    unsigned int woff[113];
    for (int i = 0; i < 113; ++i) woff[i] = 0u;
    unsigned int cur = 0;
    WConvTab wtab;
    for (int i = 0; i < 46; ++i) {
        const WInfo& w = kWTab[i];
        wtab.w[i].src = fin[w.idx];
        wtab.w[i].dstOff = cur;
        wtab.w[i].N = w.N; wtab.w[i].K = w.K; wtab.w[i].Kp = w.Kp;
        woff[w.idx] = cur;
        cur += (unsigned int)(w.N * w.Kp);
    }
    // f32 qkvcls region directly after bf16 weights (cur*2 bytes, 16B aligned)
    size_t qcByteOff = ((size_t)cur * 2 + 15) & ~(size_t)15;
    float* qc = (float*)((char*)d_ws + qcByteOff);

    AggClsTab ctab;
    ctab.a[0] = { fin[4],   fin[8],   fin[9],   64,  0    };
    ctab.a[1] = { fin[21],  fin[25],  fin[26],  64,  192  };
    ctab.a[2] = { fin[70],  fin[74],  fin[75],  128, 384  };
    ctab.a[3] = { fin[81],  fin[85],  fin[86],  128, 768  };
    ctab.a[4] = { fin[92],  fin[96],  fin[97],  128, 1152 };
    ctab.a[5] = { fin[103], fin[107], fin[108], 128, 1536 };

    // --- MainArgs ---
    MainArgs M;
    M.x   = fin[0];
    M.out = (float*)d_out;
    M.clfW = fin[111]; M.clfB = fin[112];

    auto wb = [&](int i) -> const bf16* { return wsbf + woff[i]; };

    // sub-aligner tokens: ip.flags [rb, df, mf], tcp.flags [syn, ack, fin, rst]
    const int   subWIdx[7] = {16, 12, 14, 35, 29, 31, 33};
    const int   subOffV[7] = {528, 544, 560, 912, 928, 944, 960};
    for (int t = 0; t < 7; ++t) {
        M.subW[t]   = wb(subWIdx[t]);
        M.subB[t]   = fin[subWIdx[t] + 1];
        M.subOff[t] = subOffV[t];
    }
    // sub aggs
    M.subAgg[0] = { wb(4), wb(3), wb(1), wb(2),
                    fin[8], fin[7], fin[5], fin[6], fin[11], fin[10], fin[9], qc + 0 };
    M.subAgg[1] = { wb(21), wb(20), wb(18), wb(19),
                    fin[25], fin[24], fin[22], fin[23], fin[28], fin[27], fin[26], qc + 192 };

    // field aligner tokens: eth[dst,src,type], ip[ver,hdr,flags,src,dst,proto],
    // tcp[srcport,dstport,flags,window,seq,ack]
    const int fWIdx[15] = {37,39,41, 53,47,45,51,43,49, 63,57,59,65,61,55};
    const int fOffV[15] = {0,64,128, 192,256,-1,336,400,464, 576,640,-1,720,784,848};
    for (int t = 0; t < 15; ++t) {
        M.fW[t]   = wb(fWIdx[t]);
        M.fB[t]   = fin[fWIdx[t] + 1];
        M.fOff[t] = fOffV[t];
    }
    // proto + final aggs (leaf layout within agg: W1,W2,Wo,Wqkv,b1,b2,bo,bqkv,cls,ln_b,ln_g)
    const int aggBase[4] = {67, 78, 89, 100};
    const int qcOff[4]   = {384, 768, 1152, 1536};
    for (int a = 0; a < 4; ++a) {
        int b = aggBase[a];
        AggP p = { wb(b+3), wb(b+2), wb(b+0), wb(b+1),
                   fin[b+7], fin[b+6], fin[b+4], fin[b+5],
                   fin[b+10], fin[b+9], fin[b+8], qc + qcOff[a] };
        if (a < 3) M.protoAgg[a] = p; else M.finalAgg = p;
    }

    // --- launches ---
    hipFuncSetAttribute((const void*)k_fused,
                        hipFuncAttributeMaxDynamicSharedMemorySize, SMEM_BYTES);

    k_convert_weights<<<46, 256, 0, stream>>>(wtab, wsbf);
    k_qkvcls<<<6, 128, 0, stream>>>(ctab, qc);

    const int batch = in_sizes[0] / 976;          // 32768
    const int tiles = batch / 16;                 // 2048 workgroups, 4 waves each
    k_fused<<<tiles, BLOCK, SMEM_BYTES, stream>>>(M);
}
